// Head_32847909880278
// MI455X (gfx1250) — matrix operand order repaired and checked by reading, not compiled, not run on hardware
//
#include <hip/hip_runtime.h>

// Problem constants (match reference)
#define BATCH 4
#define SEQ   4096
#define EMB   1024
#define HD    64
#define BT    (BATCH * SEQ)   // 16384

typedef __attribute__((ext_vector_type(16))) _Float16 v16h;
typedef __attribute__((ext_vector_type(8)))  _Float16 v8h;
typedef __attribute__((ext_vector_type(8)))  float    v8f;
typedef __attribute__((ext_vector_type(4)))  int      v4i;

#if defined(__gfx1250__) && __has_builtin(__builtin_amdgcn_global_load_async_to_lds_b128) && __has_builtin(__builtin_amdgcn_s_wait_asynccnt)
#define USE_ASYNC 1
#else
#define USE_ASYNC 0
#endif

__device__ __forceinline__ v16h cat8(v8h lo, v8h hi) {
    return __builtin_shufflevector(lo, hi, 0,1,2,3,4,5,6,7,8,9,10,11,12,13,14,15);
}

// A-fragment (16x32 f16): lane L<16 holds row M=L, K {0..7, 16..23}; lane>=16 K {8..15, 24..31}.
// p must already point at row_base + k0 + (lane>>4)*8 ; second chunk is +16 elements.
__device__ __forceinline__ v16h load_frag_a_h(const _Float16* p) {
    return cat8(*(const v8h*)p, *(const v8h*)(p + 16));
}

// Same A-fragment but source data is fp32 (convert in registers).
__device__ __forceinline__ v16h load_frag_a_f32(const float* p) {
    const float4 a0 = *(const float4*)(p);
    const float4 a1 = *(const float4*)(p + 4);
    const float4 b0 = *(const float4*)(p + 16);
    const float4 b1 = *(const float4*)(p + 20);
    v16h r = { (_Float16)a0.x,(_Float16)a0.y,(_Float16)a0.z,(_Float16)a0.w,
               (_Float16)a1.x,(_Float16)a1.y,(_Float16)a1.z,(_Float16)a1.w,
               (_Float16)b0.x,(_Float16)b0.y,(_Float16)b0.z,(_Float16)b0.w,
               (_Float16)b1.x,(_Float16)b1.y,(_Float16)b1.z,(_Float16)b1.w };
    return r;
}

// B-fragment (32x16 f16): lane L<16 holds col N=L, K {0..7, 16..23}; lane>=16 col N=L-16, K {8..15, 24..31}.
// p must already point at col_row_base + k0 + (lane>>4)*8 ; second chunk is +16 elements.
__device__ __forceinline__ v16h load_frag_b_h(const _Float16* p) {
    return cat8(*(const v8h*)p, *(const v8h*)(p + 16));
}

__device__ __forceinline__ v8f wmma16(v16h a, v16h b, v8f c) {
    return __builtin_amdgcn_wmma_f32_16x16x32_f16(false, a, false, b, (short)0, c, false, false);
}

__device__ __forceinline__ float rowmax16(float v) {
    v = fmaxf(v, __shfl_xor(v, 1, 32));
    v = fmaxf(v, __shfl_xor(v, 2, 32));
    v = fmaxf(v, __shfl_xor(v, 4, 32));
    v = fmaxf(v, __shfl_xor(v, 8, 32));
    return v;
}
__device__ __forceinline__ float rowsum16(float v) {
    v += __shfl_xor(v, 1, 32);
    v += __shfl_xor(v, 2, 32);
    v += __shfl_xor(v, 4, 32);
    v += __shfl_xor(v, 8, 32);
    return v;
}

__device__ __forceinline__ void lds_copy16(_Float16* lds_dst, const _Float16* gsrc) {
#if USE_ASYNC
    __builtin_amdgcn_global_load_async_to_lds_b128(
        (__attribute__((address_space(1))) v4i*)(void*)const_cast<_Float16*>(gsrc),
        (__attribute__((address_space(3))) v4i*)(void*)lds_dst, 0, 0);
#else
    *(v8h*)lds_dst = *(const v8h*)gsrc;
#endif
}

__device__ __forceinline__ void wait_async_done() {
#if USE_ASYNC
    __builtin_amdgcn_s_wait_asynccnt(0);
#endif
}

// ---------------- Kernel 1: transpose/convert weights to f16, layout [mat][n][k] ----------------
__global__ __launch_bounds__(256) void head_prep_w(const float* __restrict__ Wq,
                                                   const float* __restrict__ Wk,
                                                   const float* __restrict__ Wv,
                                                   _Float16* __restrict__ Wt) {
    int i = blockIdx.x * blockDim.x + threadIdx.x;       // [mat][n][k] flat, 3*64*1024
    if (i >= 3 * HD * EMB) return;
    int mat = i >> 16;            // /(64*1024)
    int rem = i & 65535;
    int n = rem >> 10;            // /1024
    int k = rem & 1023;
    const float* W = (mat == 0) ? Wq : (mat == 1) ? Wk : Wv;   // W is [C][D] row-major
    Wt[i] = (_Float16)W[k * HD + n];
}

// ---------------- Kernel 2: QKV projection via WMMA ----------------
// grid: BT/64 blocks of 128 threads; wave w owns 16-row M-tile (blockIdx.x*4 + w)*16.
// Each wave computes 12 16x16 tiles: {Q,K,V} x 4 N-tiles, K-loop over C in steps of 32.
__global__ __launch_bounds__(128) void head_qkv_proj(const float* __restrict__ X,
                                                     const _Float16* __restrict__ Wt,
                                                     _Float16* __restrict__ Qh,
                                                     _Float16* __restrict__ Kh,
                                                     _Float16* __restrict__ Vt) {
    const int lane = threadIdx.x & 31;
    const int wave = threadIdx.x >> 5;
    const int m0 = (blockIdx.x * 4 + wave) * 16;
    const int colL = lane & 15;
    const int half = lane >> 4;

    v8f acc[12];
#pragma unroll
    for (int t = 0; t < 12; ++t) acc[t] = (v8f)0.0f;

    const float* xrow = X + (size_t)(m0 + colL) * EMB + half * 8;

#pragma unroll 2
    for (int k0 = 0; k0 < EMB; k0 += 32) {
        v16h a = load_frag_a_f32(xrow + k0);
#pragma unroll
        for (int mat = 0; mat < 3; ++mat) {
            const _Float16* wb = Wt + mat * (HD * EMB) + (size_t)colL * EMB + k0 + half * 8;
#pragma unroll
            for (int ni = 0; ni < 4; ++ni) {
                v16h b = load_frag_b_h(wb + ni * 16 * EMB);
                acc[mat * 4 + ni] = wmma16(a, b, acc[mat * 4 + ni]);
            }
        }
    }

    // C/D layout: lane holds col N=lane&15 of rows (lane>>4)*8 + r
    const int rbase = half * 8;
#pragma unroll
    for (int ni = 0; ni < 4; ++ni) {
        int col = ni * 16 + colL;
#pragma unroll
        for (int r = 0; r < 8; ++r) {
            int row = m0 + rbase + r;
            Qh[(size_t)row * HD + col] = (_Float16)acc[0 * 4 + ni][r];
            Kh[(size_t)row * HD + col] = (_Float16)acc[1 * 4 + ni][r];
            int b = row >> 12;             // /SEQ
            int t = row & (SEQ - 1);
            Vt[((size_t)b * HD + col) * SEQ + t] = (_Float16)acc[2 * 4 + ni][r];
        }
    }
}

// ---------------- Kernel 3: flash attention, async-LDS double-buffered K/V ----------------
// grid (SEQ/64, BATCH), 128 threads. Wave w owns rows r0 + w*16 .. +16.
__global__ __launch_bounds__(128) void head_flash_attn(const _Float16* __restrict__ Qh,
                                                       const _Float16* __restrict__ Kh,
                                                       const _Float16* __restrict__ Vt,
                                                       float* __restrict__ out) {
    __shared__ _Float16 Kl[2][64][72];   // K block  [s][d], padded rows (144B stride)
    __shared__ _Float16 Vl[2][64][72];   // V block  [d][t], padded rows
    __shared__ _Float16 Pl[4][16][72];   // per-wave P staging (C-layout -> A-layout)

    const int tid  = threadIdx.x;
    const int lane = tid & 31;
    const int wave = tid >> 5;
    const int b    = blockIdx.y;
    const int r0   = blockIdx.x * 64;
    const int m0   = r0 + wave * 16;
    const int colL = lane & 15;
    const int half = lane >> 4;
    const int rbase = half * 8;

    const _Float16* Qb = Qh + (size_t)b * SEQ * HD;
    const _Float16* Kb = Kh + (size_t)b * SEQ * HD;
    const _Float16* Vb = Vt + (size_t)b * HD * SEQ;

    // Q fragments are loop-invariant per wave: preload both K-steps.
    v16h qa[2];
#pragma unroll
    for (int kk2 = 0; kk2 < 2; ++kk2)
        qa[kk2] = load_frag_a_h(Qb + (size_t)(m0 + colL) * HD + kk2 * 32 + half * 8);

    v8f oacc[4];
#pragma unroll
    for (int d = 0; d < 4; ++d) oacc[d] = (v8f)0.0f;
    float mrow[8], lrow[8];
#pragma unroll
    for (int r = 0; r < 8; ++r) { mrow[r] = -3.0e38f; lrow[r] = 0.0f; }

    const float scale = 0.03125f;   // EMB^-0.5 = 1/32

    // Cooperative async copy of one 64x64 K block + 64x64 V block into LDS buffer `buf`.
    // 1024 16-byte chunks over 128 threads = 8 chunks/thread (4 iterations x {K,V}).
    auto issue_copy = [&](int buf, int c0) {
#pragma unroll
        for (int j = 0; j < 4; ++j) {
            int c   = j * 128 + tid;       // 0..511
            int row = c >> 3;
            int ch  = (c & 7) * 8;         // element offset within row
            lds_copy16(&Kl[buf][row][ch], Kb + (size_t)(c0 + row) * HD + ch);
            lds_copy16(&Vl[buf][row][ch], Vb + (size_t)row * SEQ + c0 + ch);
        }
    };

    issue_copy(0, 0);

    for (int c0 = 0; c0 <= r0; c0 += 64) {
        const int cur = (c0 >> 6) & 1;

        wait_async_done();      // this wave's copies for buf[cur] landed in LDS
        __syncthreads();        // all waves' copies landed; all done reading buf[cur^1]

        if (c0 + 64 <= r0) issue_copy(cur ^ 1, c0 + 64);   // prefetch next block

        // ---- S = Q @ K^T (64 cols) from LDS ----
        v8f s[4];
#pragma unroll
        for (int ni = 0; ni < 4; ++ni) s[ni] = (v8f)0.0f;
#pragma unroll
        for (int kk2 = 0; kk2 < 2; ++kk2) {
#pragma unroll
            for (int ni = 0; ni < 4; ++ni) {
                v16h kb = load_frag_b_h(&Kl[cur][ni * 16 + colL][kk2 * 32 + half * 8]);
                s[ni] = wmma16(qa[kk2], kb, s[ni]);
            }
        }

        // ---- scale + causal mask (diagonal block only) ----
        const bool diag = (c0 == r0);
#pragma unroll
        for (int ni = 0; ni < 4; ++ni) {
#pragma unroll
            for (int r = 0; r < 8; ++r) {
                float v = s[ni][r] * scale;
                if (diag) {
                    int cc = c0 + ni * 16 + colL;
                    int rr = m0 + rbase + r;
                    if (cc > rr) v = -3.0e38f;
                }
                s[ni][r] = v;
            }
        }

        // ---- online softmax ----
#pragma unroll
        for (int r = 0; r < 8; ++r) {
            float v = fmaxf(fmaxf(s[0][r], s[1][r]), fmaxf(s[2][r], s[3][r]));
            float mnew = fmaxf(mrow[r], rowmax16(v));
            float alpha = __expf(mrow[r] - mnew);
            float rs = 0.0f;
#pragma unroll
            for (int ni = 0; ni < 4; ++ni) {
                float p = __expf(s[ni][r] - mnew);
                s[ni][r] = p;
                rs += p;
            }
            rs = rowsum16(rs);
            lrow[r] = lrow[r] * alpha + rs;
            mrow[r] = mnew;
#pragma unroll
            for (int d = 0; d < 4; ++d) oacc[d][r] *= alpha;
        }

        // ---- P: C-layout -> LDS -> A-layout ----
#pragma unroll
        for (int ni = 0; ni < 4; ++ni)
#pragma unroll
            for (int r = 0; r < 8; ++r)
                Pl[wave][rbase + r][ni * 16 + colL] = (_Float16)s[ni][r];
        __syncthreads();   // uniform across block (same trip count all waves)

        // ---- O += P @ V from LDS ----
#pragma unroll
        for (int kk2 = 0; kk2 < 2; ++kk2) {
            const _Float16* pp = &Pl[wave][colL][kk2 * 32 + half * 8];
            v16h pa = cat8(*(const v8h*)pp, *(const v8h*)(pp + 16));
#pragma unroll
            for (int di = 0; di < 4; ++di) {
                v16h vb = load_frag_b_h(&Vl[cur][di * 16 + colL][kk2 * 32 + half * 8]);
                oacc[di] = wmma16(pa, vb, oacc[di]);
            }
        }
    }

    // ---- epilogue: normalize and store fp32 ----
    float* ob = out + ((size_t)b * SEQ + m0) * HD;
#pragma unroll
    for (int di = 0; di < 4; ++di) {
        int col = di * 16 + colL;
#pragma unroll
        for (int r = 0; r < 8; ++r) {
            ob[(size_t)(rbase + r) * HD + col] = oacc[di][r] / lrow[r];
        }
    }
}

extern "C" void kernel_launch(void* const* d_in, const int* in_sizes, int n_in,
                              void* d_out, int out_size, void* d_ws, size_t ws_size,
                              hipStream_t stream) {
    const float* X  = (const float*)d_in[0];
    const float* Wq = (const float*)d_in[1];
    const float* Wk = (const float*)d_in[2];
    const float* Wv = (const float*)d_in[3];
    float* out = (float*)d_out;

    // workspace layout (f16): Wt [3][64][1024], Qh [BT][64], Kh [BT][64], Vt [B][64][SEQ]
    char* ws = (char*)d_ws;
    const size_t WT_BYTES = (size_t)3 * HD * EMB * 2;   // 393216
    const size_t QK_BYTES = (size_t)BT * HD * 2;        // 2097152
    _Float16* Wt = (_Float16*)(ws);
    _Float16* Qh = (_Float16*)(ws + WT_BYTES);
    _Float16* Kh = (_Float16*)(ws + WT_BYTES + QK_BYTES);
    _Float16* Vt = (_Float16*)(ws + WT_BYTES + 2 * QK_BYTES);

    head_prep_w<<<(3 * HD * EMB + 255) / 256, 256, 0, stream>>>(Wq, Wk, Wv, Wt);
    head_qkv_proj<<<BT / 64, 128, 0, stream>>>(X, Wt, Qh, Kh, Vt);
    head_flash_attn<<<dim3(SEQ / 64, BATCH), 128, 0, stream>>>(Qh, Kh, Vt, out);
}
